// NaiveRNN_6390911336712
// MI455X (gfx1250) — matrix operand order repaired and checked by reading, not compiled, not run on hardware
//
#include <hip/hip_runtime.h>

// Problem dims (fixed by reference)
#define BB   128
#define LL   512
#define DIN  512
#define DH   1024
#define DOUT 512
#define NBLK 64          // workgroups in recurrent kernel (grid barrier count)
#define PAD  1024        // element padding for harmless prefetch overreads

typedef __attribute__((ext_vector_type(16))) __bf16 v16bf;
typedef __attribute__((ext_vector_type(8)))  __bf16 v8bf;
typedef __attribute__((ext_vector_type(4)))  __bf16 v4bf;
typedef __attribute__((ext_vector_type(4)))  float  v4f;
typedef __attribute__((ext_vector_type(8)))  float  v8f;

// ---------------------------------------------------------------------------
// Fragment layouts (CDNA5 wave32 WMMA 16x16x32 bf16):
//  A (16x32 MxK): lane l (m=l&15, h=l>>4): two 16B chunks @ k0+8h, k0+16+8h
//  B (32x16 KxN): pre-swizzled global: [kb][nb][lane][16] -> 32B/lane
//  C/D (16x16 f32): lane l (n=l&15, mh=l>>4), r=0..7 -> M = 8*mh + r
// ---------------------------------------------------------------------------

__device__ __forceinline__ v16bf load_afrag(const __bf16* row, int k0,
                                            int half) {
  v8bf lo = *(const v8bf*)(row + k0 + half * 8);
  v8bf hi = *(const v8bf*)(row + k0 + 16 + half * 8);
  v16bf a;
  #pragma unroll
  for (int j = 0; j < 8; ++j) { a[j] = lo[j]; a[8 + j] = hi[j]; }
  return a;
}

#define WMMA_BF16(A, B, C)                                                   \
  __builtin_amdgcn_wmma_f32_16x16x32_bf16(false, (A), false, (B), (short)0,  \
                                          (C), false, false)

// ---- one-time prep -------------------------------------------------------

__global__ __launch_bounds__(256) void cvt_x_kernel(
    const float* __restrict__ x, __bf16* __restrict__ xb) {
  int i = blockIdx.x * blockDim.x + threadIdx.x;   // over total/4
  v4f v = *(const v4f*)(x + (size_t)i * 4);
  v4bf o;
  o[0] = (__bf16)v[0]; o[1] = (__bf16)v[1];
  o[2] = (__bf16)v[2]; o[3] = (__bf16)v[3];
  *(v4bf*)(xb + (size_t)i * 4) = o;
}

// W [K][N] f32 -> bf16 B-fragment order: dst[((kb*(N/16)+nb)*32+ln)*16+j]
__global__ __launch_bounds__(256) void swz_weight_kernel(
    const float* __restrict__ W, __bf16* __restrict__ dst, int K, int N) {
  int total = K * N;
  for (int idx = blockIdx.x * blockDim.x + threadIdx.x; idx < total;
       idx += gridDim.x * blockDim.x) {
    int k = idx / N, n = idx % N;                 // coalesced along n
    int kb = k >> 5, kk = k & 31;
    int ln = (n & 15) + (((kk >> 3) & 1) << 4);
    dst[((size_t)(kb * (N >> 4) + (n >> 4)) * 32 + ln) * 16 + ((kk & 7) + ((kk >> 4) << 3))] =
        (__bf16)W[idx];
  }
}

__global__ void init_ws_kernel(__bf16* h0, unsigned* arrive) {
  int i = blockIdx.x * blockDim.x + threadIdx.x;
  if (i < BB * DH + PAD) h0[i] = (__bf16)0.0f;
  if (i < LL) arrive[i] = 0u;
}

// ---- xi = x @ W_i2h + b_i2h, stored bf16 [L][B][DH] ----------------------
// LDS-free, register-double-buffered. Wave tile: 32 rows x 64 cols.
__global__ __launch_bounds__(256) void xi_gemm_kernel(
    const __bf16* __restrict__ xb, const __bf16* __restrict__ Wswz,
    const float* __restrict__ bias, __bf16* __restrict__ xi_out) {
  const int lane = threadIdx.x & 31, wave = threadIdx.x >> 5;
  const int m0 = blockIdx.x * 256 + wave * 32;  // 8 waves x 32 rows
  const int nb0 = blockIdx.y * 4;               // 4 x 16 = 64 cols
  const int nl = lane & 15, half = lane >> 4;
  constexpr int KB = DIN / 32;
  constexpr int kbStride = (DH >> 4) * 512;     // elements per kb in Wswz

  const __bf16* ar0 = xb + (size_t)(m0 + nl) * DIN;
  const __bf16* ar1 = xb + (size_t)(m0 + 16 + nl) * DIN;
  const __bf16* bbase = Wswz + (size_t)nb0 * 512 + lane * 16;

  v8f acc[2][4] = {};
  v16bf a0[2], a1[2], b0[4], b1[4];

  a0[0] = load_afrag(ar0, 0, half);
  a0[1] = load_afrag(ar1, 0, half);
  #pragma unroll
  for (int nb = 0; nb < 4; ++nb) b0[nb] = *(const v16bf*)(bbase + nb * 512);

  for (int kb = 0; kb < KB; kb += 2) {
    // prefetch kb+1 into set1
    a1[0] = load_afrag(ar0, (kb + 1) * 32, half);
    a1[1] = load_afrag(ar1, (kb + 1) * 32, half);
    #pragma unroll
    for (int nb = 0; nb < 4; ++nb)
      b1[nb] = *(const v16bf*)(bbase + (size_t)(kb + 1) * kbStride + nb * 512);
    #pragma unroll
    for (int mb = 0; mb < 2; ++mb)
      #pragma unroll
      for (int nb = 0; nb < 4; ++nb)
        acc[mb][nb] = WMMA_BF16(a0[mb], b0[nb], acc[mb][nb]);
    // prefetch kb+2 into set0 (padded overread on last iteration)
    a0[0] = load_afrag(ar0, (kb + 2) * 32, half);
    a0[1] = load_afrag(ar1, (kb + 2) * 32, half);
    #pragma unroll
    for (int nb = 0; nb < 4; ++nb)
      b0[nb] = *(const v16bf*)(bbase + (size_t)(kb + 2) * kbStride + nb * 512);
    #pragma unroll
    for (int mb = 0; mb < 2; ++mb)
      #pragma unroll
      for (int nb = 0; nb < 4; ++nb)
        acc[mb][nb] = WMMA_BF16(a1[mb], b1[nb], acc[mb][nb]);
  }

  #pragma unroll
  for (int mb = 0; mb < 2; ++mb) {
    int m_base = m0 + mb * 16 + half * 8;       // 8-aligned: same b for r=0..7
    int bidx = m_base >> 9, tstep = m_base & (LL - 1);
    #pragma unroll
    for (int nb = 0; nb < 4; ++nb) {
      int n = (nb0 + nb) * 16 + nl;
      float bv = bias[n];
      __bf16* dst = xi_out + ((size_t)tstep * BB + bidx) * DH + n;
      #pragma unroll
      for (int r = 0; r < 8; ++r)
        dst[(size_t)r * BB * DH] = (__bf16)(acc[mb][nb][r] + bv);
    }
  }
}

// ---- persistent recurrence ----------------------------------------------
// 64 WGs x 128 threads (4 waves x 32 rows). 16 cols of W_h2h in LDS.
__global__ __launch_bounds__(128) void rnn_steps_kernel(
    const __bf16* __restrict__ Whswz, const float* __restrict__ bh,
    const __bf16* __restrict__ xi, __bf16* __restrict__ h0buf,
    __bf16* __restrict__ h1buf, unsigned* __restrict__ arrive) {
  __shared__ __align__(32) __bf16 Wlds[33 * 512];  // 32 kb blocks + pad
  const int tid = threadIdx.x, lane = tid & 31, wave = tid >> 5;
  const int nb0 = blockIdx.x;                   // n0 = nb0*16

  // bulk-copy fragment-order W slice into LDS (16B moves)
  for (int c = tid; c < 2048; c += 128) {       // 2048 x 16B = 32KB
    int kb = c >> 6, within = (c & 63) * 8;
    *(v8bf*)&Wlds[kb * 512 + within] =
        *(const v8bf*)&Whswz[((size_t)(kb * (DH >> 4) + nb0)) * 512 + within];
  }
  __syncthreads();

  const int nl = lane & 15, half = lane >> 4;
  const float bhv = bh[nb0 * 16 + nl];
  const int m0 = wave * 32;                     // 4 waves x 32 rows = B=128
  const __bf16* blds = &Wlds[lane * 16];
  __bf16* bufs[2] = {h0buf, h1buf};

  for (int t = 0; t < LL; ++t) {
    const __bf16* hsrc = bufs[t & 1];
    __bf16* hdst = bufs[(t + 1) & 1];
    const __bf16* ar0 = hsrc + (size_t)(m0 + nl) * DH;
    const __bf16* ar1 = hsrc + (size_t)(m0 + 16 + nl) * DH;

    v8f acc0 = {}, acc1 = {};
    v16bf a0[2], a1[2], b0, b1;
    a0[0] = load_afrag(ar0, 0, half);
    a0[1] = load_afrag(ar1, 0, half);
    b0 = *(const v16bf*)(blds);
    for (int kb = 0; kb < 32; kb += 2) {
      a1[0] = load_afrag(ar0, (kb + 1) * 32, half);
      a1[1] = load_afrag(ar1, (kb + 1) * 32, half);
      b1 = *(const v16bf*)(blds + (kb + 1) * 512);
      acc0 = WMMA_BF16(a0[0], b0, acc0);
      acc1 = WMMA_BF16(a0[1], b0, acc1);
      a0[0] = load_afrag(ar0, (kb + 2) * 32, half);   // padded overread at end
      a0[1] = load_afrag(ar1, (kb + 2) * 32, half);
      b0 = *(const v16bf*)(blds + (kb + 2) * 512);
      acc0 = WMMA_BF16(a1[0], b1, acc0);
      acc1 = WMMA_BF16(a1[1], b1, acc1);
    }

    const __bf16* xit = xi + (size_t)t * BB * DH + nb0 * 16 + nl;
    __bf16* hd = hdst + nb0 * 16 + nl;
    #pragma unroll
    for (int r = 0; r < 8; ++r) {
      int m = m0 + half * 8 + r;
      float v0 = acc0[r] + (float)xit[(size_t)m * DH] + bhv;
      hd[(size_t)m * DH] = (__bf16)tanhf(v0);
      int m2 = m + 16;
      float v1 = acc1[r] + (float)xit[(size_t)m2 * DH] + bhv;
      hd[(size_t)m2 * DH] = (__bf16)tanhf(v1);
    }
    // device-scope barrier for step t
    __syncthreads();
    if (tid == 0) {
      __threadfence();
      atomicAdd(&arrive[t], 1u);
      while (__hip_atomic_load(&arrive[t], __ATOMIC_ACQUIRE,
                               __HIP_MEMORY_SCOPE_AGENT) < (unsigned)NBLK) {
        __builtin_amdgcn_s_sleep(1);
      }
    }
    __syncthreads();
  }
}

// ---- out = h_final @ W_out + b_out (f32), LDS-free -----------------------
// 4 waves x 32 rows x 64 cols, register-double-buffered.
__global__ __launch_bounds__(128) void out_gemm_kernel(
    const __bf16* __restrict__ h, const __bf16* __restrict__ Wswz,
    const float* __restrict__ bo, float* __restrict__ out) {
  const int lane = threadIdx.x & 31, wave = threadIdx.x >> 5;
  const int nb0 = blockIdx.x * 4;
  const int nl = lane & 15, half = lane >> 4;
  const int m0 = wave * 32;
  constexpr int KB = DH / 32;
  constexpr int kbStride = (DOUT >> 4) * 512;

  const __bf16* ar0 = h + (size_t)(m0 + nl) * DH;
  const __bf16* ar1 = h + (size_t)(m0 + 16 + nl) * DH;
  const __bf16* bbase = Wswz + (size_t)nb0 * 512 + lane * 16;

  v8f acc[2][4] = {};
  v16bf a0[2], a1[2], b0[4], b1[4];
  a0[0] = load_afrag(ar0, 0, half);
  a0[1] = load_afrag(ar1, 0, half);
  #pragma unroll
  for (int nb = 0; nb < 4; ++nb) b0[nb] = *(const v16bf*)(bbase + nb * 512);

  for (int kb = 0; kb < KB; kb += 2) {
    a1[0] = load_afrag(ar0, (kb + 1) * 32, half);
    a1[1] = load_afrag(ar1, (kb + 1) * 32, half);
    #pragma unroll
    for (int nb = 0; nb < 4; ++nb)
      b1[nb] = *(const v16bf*)(bbase + (size_t)(kb + 1) * kbStride + nb * 512);
    #pragma unroll
    for (int mb = 0; mb < 2; ++mb)
      #pragma unroll
      for (int nb = 0; nb < 4; ++nb)
        acc[mb][nb] = WMMA_BF16(a0[mb], b0[nb], acc[mb][nb]);
    a0[0] = load_afrag(ar0, (kb + 2) * 32, half);
    a0[1] = load_afrag(ar1, (kb + 2) * 32, half);
    #pragma unroll
    for (int nb = 0; nb < 4; ++nb)
      b0[nb] = *(const v16bf*)(bbase + (size_t)(kb + 2) * kbStride + nb * 512);
    #pragma unroll
    for (int mb = 0; mb < 2; ++mb)
      #pragma unroll
      for (int nb = 0; nb < 4; ++nb)
        acc[mb][nb] = WMMA_BF16(a1[mb], b1[nb], acc[mb][nb]);
  }

  #pragma unroll
  for (int mb = 0; mb < 2; ++mb) {
    int m_base = m0 + mb * 16 + half * 8;
    #pragma unroll
    for (int nb = 0; nb < 4; ++nb) {
      int n = (nb0 + nb) * 16 + nl;
      float bv = bo[n];
      float* dst = out + (size_t)m_base * DOUT + n;
      #pragma unroll
      for (int r = 0; r < 8; ++r) dst[r * DOUT] = acc[mb][nb][r] + bv;
    }
  }
}

extern "C" void kernel_launch(void* const* d_in, const int* in_sizes, int n_in,
                              void* d_out, int out_size, void* d_ws,
                              size_t ws_size, hipStream_t stream) {
  const float* x     = (const float*)d_in[0];
  const float* W_i2h = (const float*)d_in[1];
  const float* b_i2h = (const float*)d_in[2];
  const float* W_h2h = (const float*)d_in[3];
  const float* b_h2h = (const float*)d_in[4];
  const float* W_out = (const float*)d_in[5];
  const float* b_out = (const float*)d_in[6];
  float* out = (float*)d_out;

  // Workspace layout (PAD elements after over-readable buffers):
  //   xi bf16 [L][B][DH] 128MB | x bf16 64MB | Wi/Wh/Wo swz 1/2/1MB
  //   h ping-pong 2x256KB | arrive 2KB
  char* ws = (char*)d_ws;
  size_t off = 0;
  __bf16* xi_bf = (__bf16*)(ws + off); off += ((size_t)LL * BB * DH + PAD) * 2;
  __bf16* x_bf  = (__bf16*)(ws + off); off += ((size_t)BB * LL * DIN + PAD) * 2;
  __bf16* Wi_swz = (__bf16*)(ws + off); off += ((size_t)DIN * DH + PAD) * 2;
  __bf16* Wh_swz = (__bf16*)(ws + off); off += ((size_t)DH * DH + PAD) * 2;
  __bf16* Wo_swz = (__bf16*)(ws + off); off += ((size_t)DH * DOUT + PAD) * 2;
  __bf16* h0 = (__bf16*)(ws + off); off += ((size_t)BB * DH + PAD) * 2;
  __bf16* h1 = (__bf16*)(ws + off); off += ((size_t)BB * DH + PAD) * 2;
  unsigned* arrive = (unsigned*)(ws + off);

  init_ws_kernel<<<(BB * DH + PAD + 255) / 256, 256, 0, stream>>>(h0, arrive);
  cvt_x_kernel<<<(BB * LL * DIN / 4) / 256, 256, 0, stream>>>(x, x_bf);
  swz_weight_kernel<<<512, 256, 0, stream>>>(W_i2h, Wi_swz, DIN, DH);
  swz_weight_kernel<<<512, 256, 0, stream>>>(W_h2h, Wh_swz, DH, DH);
  swz_weight_kernel<<<512, 256, 0, stream>>>(W_out, Wo_swz, DH, DOUT);

  dim3 xi_grid((BB * LL) / 256, DH / 64);
  xi_gemm_kernel<<<xi_grid, 256, 0, stream>>>(x_bf, Wi_swz, b_i2h, xi_bf);

  rnn_steps_kernel<<<NBLK, 128, 0, stream>>>(Wh_swz, b_h2h, xi_bf, h0, h1,
                                             arrive);

  out_gemm_kernel<<<DOUT / 64, 128, 0, stream>>>(h0, Wo_swz, b_out, out);
}
